// CrossAttentionBridge_33921651704567
// MI455X (gfx1250) — compile-verified, matrix operand order checked by reading, not run on hardware
//
#include <hip/hip_runtime.h>
#include <hip/hip_bf16.h>

#define E_DIM   1024
#define VITD    512
#define B_ROWS  16384

#ifndef USE_ASYNC_COPY
#define USE_ASYNC_COPY 1
#endif

typedef __attribute__((ext_vector_type(16))) _Float16 v16h;
typedef __attribute__((ext_vector_type(8)))  _Float16 v8h;
typedef __attribute__((ext_vector_type(8)))  float    v8f;

union Frag { v16h v; v8h h[2]; };

// ---------------------------------------------------------------------------
// async raw copy: 16 bytes global -> LDS, tracked by ASYNCcnt
// ---------------------------------------------------------------------------
__device__ __forceinline__ void async_b128(unsigned int ldsOff, const void* g)
{
#if USE_ASYNC_COPY
    asm volatile("global_load_async_to_lds_b128 %0, %1, off"
                 :: "v"(ldsOff), "v"(g) : "memory");
#else
    // fallback: plain load + ds store (compiler-generated)
    *(v8h*)(((char*)0) + 0) = *(const v8h*)g; // never used
#endif
}

__device__ __forceinline__ void wait_async_le8()
{
#if USE_ASYNC_COPY
#if __has_builtin(__builtin_amdgcn_s_wait_asynccnt)
    __builtin_amdgcn_s_wait_asynccnt(8);
#else
    asm volatile("s_wait_asynccnt 8" ::: "memory");
#endif
#endif
}

__device__ __forceinline__ void wait_async_le0()
{
#if USE_ASYNC_COPY
#if __has_builtin(__builtin_amdgcn_s_wait_asynccnt)
    __builtin_amdgcn_s_wait_asynccnt(0);
#else
    asm volatile("s_wait_asynccnt 0" ::: "memory");
#endif
#endif
}

// ---------------------------------------------------------------------------
// Main GEMM on pre-split f16 hi/lo planes:
//   C[M,N] = A @ B^T + bias,  A planes [M][K], B planes [N][K] (k contiguous)
// fp32 accuracy via hi*hi + hi*lo + lo*hi (~22-bit mantissa).
// 128 threads = 4 waves; 64x64 block tile; BK=32; double-buffered async LDS.
// Requires M%64==0, N%64==0, K%32==0.
// ---------------------------------------------------------------------------
#define BK   32
#define LSTR 40                       // halves per LDS row (80B, 16B-aligned segs)

__global__ __launch_bounds__(128)
void wmma_gemm_split(const _Float16* __restrict__ Ahi, const _Float16* __restrict__ Alo,
                     const _Float16* __restrict__ Bhi, const _Float16* __restrict__ Blo,
                     const float* __restrict__ bias, float* __restrict__ C,
                     int M, int N, int K)
{
    // [buf][plane: 0=Ahi 1=Alo 2=Bhi 3=Blo][64 rows * LSTR halves]
    __shared__ __align__(16) _Float16 sm[2][4][64 * LSTR];

    const int tid    = threadIdx.x;
    const int lane   = tid & 31;
    const int wave   = tid >> 5;
    const int waveM  = (wave >> 1) * 32;
    const int waveN  = (wave &  1) * 32;
    const int rowSel = lane & 15;
    const int kb     = (lane >> 4) << 3;     // 0 / 8 : lanes 16..31 own K+8
    const int mOff   = (lane >> 4) << 3;

    const int tileM = blockIdx.x * 64;
    const int tileN = blockIdx.y * 64;

    const _Float16* gplane[4] = { Ahi, Alo, Bhi, Blo };

    v8f acc[2][2] = {};
    const int nch = K / BK;

    // stage chunk `ch` into buffer `buf`: 4 planes * 64 rows * 4 segs of 8 halves
    auto stage = [&](int ch, int buf) {
        const int kk = ch * BK;
        #pragma unroll
        for (int p = 0; p < 4; ++p) {
            const _Float16* src = gplane[p];
            const int rowBase   = (p < 2) ? tileM : tileN;
            const unsigned int lbase = (unsigned int)(uintptr_t)&sm[buf][p][0];
            #pragma unroll
            for (int idx = tid; idx < 256; idx += 128) {       // 2 iters
                int r = idx >> 2, s = idx & 3;
                const void* g = src + (size_t)(rowBase + r) * K + kk + s * 8;
                unsigned int loff = lbase + (unsigned int)(r * (LSTR * 2) + s * 16);
#if USE_ASYNC_COPY
                async_b128(loff, g);
#else
                *(v8h*)&sm[buf][p][r * LSTR + s * 8] = *(const v8h*)g;
#endif
            }
        }
    };

    stage(0, 0);

    for (int ch = 0; ch < nch; ++ch) {
        const int buf = ch & 1;
        if (ch + 1 < nch) { stage(ch + 1, buf ^ 1); wait_async_le8(); }
        else              { wait_async_le0(); }
        __syncthreads();                         // chunk `ch` visible to all waves

        Frag fahi[2], falo[2], fbhi[2], fblo[2];
        #pragma unroll
        for (int i = 0; i < 2; ++i) {
            int row = waveM + i * 16 + rowSel;
            const char* phi = (const char*)&sm[buf][0][0] + row * (LSTR * 2) + kb * 2;
            const char* plo = (const char*)&sm[buf][1][0] + row * (LSTR * 2) + kb * 2;
            fahi[i].h[0] = *(const v8h*)(phi);      fahi[i].h[1] = *(const v8h*)(phi + 32);
            falo[i].h[0] = *(const v8h*)(plo);      falo[i].h[1] = *(const v8h*)(plo + 32);
        }
        #pragma unroll
        for (int i = 0; i < 2; ++i) {
            int col = waveN + i * 16 + rowSel;
            const char* phi = (const char*)&sm[buf][2][0] + col * (LSTR * 2) + kb * 2;
            const char* plo = (const char*)&sm[buf][3][0] + col * (LSTR * 2) + kb * 2;
            fbhi[i].h[0] = *(const v8h*)(phi);      fbhi[i].h[1] = *(const v8h*)(phi + 32);
            fblo[i].h[0] = *(const v8h*)(plo);      fblo[i].h[1] = *(const v8h*)(plo + 32);
        }

        #pragma unroll
        for (int i = 0; i < 2; ++i) {
            #pragma unroll
            for (int j = 0; j < 2; ++j) {
                acc[i][j] = __builtin_amdgcn_wmma_f32_16x16x32_f16(
                    false, falo[i].v, false, fbhi[j].v, (short)0, acc[i][j], false, false);
                acc[i][j] = __builtin_amdgcn_wmma_f32_16x16x32_f16(
                    false, fahi[i].v, false, fblo[j].v, (short)0, acc[i][j], false, false);
                acc[i][j] = __builtin_amdgcn_wmma_f32_16x16x32_f16(
                    false, fahi[i].v, false, fbhi[j].v, (short)0, acc[i][j], false, false);
            }
        }
        __syncthreads();                         // close WAR window on other buffer
    }

    #pragma unroll
    for (int i = 0; i < 2; ++i) {
        #pragma unroll
        for (int j = 0; j < 2; ++j) {
            int ncol = tileN + waveN + j * 16 + rowSel;
            float badd = bias ? bias[ncol] : 0.0f;
            #pragma unroll
            for (int r = 0; r < 8; ++r) {
                int mrow = tileM + waveM + i * 16 + mOff + r;
                C[(size_t)mrow * N + ncol] = acc[i][j][r] + badd;
            }
        }
    }
}

// ---------------------------------------------------------------------------
// Fold GEMM (small, in-kernel split): C[M,N] = A[M,K] @ B + bias?
// B element (k,n) at B[k*sBK + n*sBN].  Same math as round 1.
// ---------------------------------------------------------------------------
__global__ __launch_bounds__(128)
void wmma_gemm_f32x3(const float* __restrict__ A, const float* __restrict__ B,
                     const float* __restrict__ bias, float* __restrict__ C,
                     int M, int N, int K, int sBK, int sBN)
{
    __shared__ float lA[64][33];
    __shared__ float lB[32][65];

    const int tid    = threadIdx.x;
    const int lane   = tid & 31;
    const int wave   = tid >> 5;
    const int waveM  = (wave >> 1) * 32;
    const int waveN  = (wave &  1) * 32;
    const int rowSel = lane & 15;
    const int kb     = (lane >> 4) << 3;
    const int mOff   = (lane >> 4) << 3;

    const int tileM = blockIdx.x * 64;
    const int tileN = blockIdx.y * 64;

    v8f acc[2][2] = {};

    for (int kk = 0; kk < K; kk += 32) {
        #pragma unroll
        for (int idx = tid; idx < 64 * 32; idx += 128) {
            int r = idx >> 5, c = idx & 31;
            lA[r][c] = A[(size_t)(tileM + r) * K + (kk + c)];
        }
        #pragma unroll
        for (int idx = tid; idx < 32 * 64; idx += 128) {
            int k = idx >> 6, n = idx & 63;
            lB[k][n] = B[(size_t)(kk + k) * sBK + (size_t)(tileN + n) * sBN];
        }
        __syncthreads();

        v16h ahi[2], alo[2], bhi[2], blo[2];
        #pragma unroll
        for (int i = 0; i < 2; ++i) {
            int r = waveM + i * 16 + rowSel;
            #pragma unroll
            for (int j = 0; j < 16; ++j) {
                int k = ((j >> 3) << 4) | (kb + (j & 7));
                float x = lA[r][k];
                _Float16 h = (_Float16)x;
                ahi[i][j] = h;  alo[i][j] = (_Float16)(x - (float)h);
            }
        }
        #pragma unroll
        for (int i = 0; i < 2; ++i) {
            int c = waveN + i * 16 + rowSel;
            #pragma unroll
            for (int j = 0; j < 16; ++j) {
                int k = ((j >> 3) << 4) | (kb + (j & 7));
                float x = lB[k][c];
                _Float16 h = (_Float16)x;
                bhi[i][j] = h;  blo[i][j] = (_Float16)(x - (float)h);
            }
        }

        #pragma unroll
        for (int i = 0; i < 2; ++i) {
            #pragma unroll
            for (int j = 0; j < 2; ++j) {
                acc[i][j] = __builtin_amdgcn_wmma_f32_16x16x32_f16(
                    false, alo[i], false, bhi[j], (short)0, acc[i][j], false, false);
                acc[i][j] = __builtin_amdgcn_wmma_f32_16x16x32_f16(
                    false, ahi[i], false, blo[j], (short)0, acc[i][j], false, false);
                acc[i][j] = __builtin_amdgcn_wmma_f32_16x16x32_f16(
                    false, ahi[i], false, bhi[j], (short)0, acc[i][j], false, false);
            }
        }
        __syncthreads();
    }

    #pragma unroll
    for (int i = 0; i < 2; ++i) {
        #pragma unroll
        for (int j = 0; j < 2; ++j) {
            int ncol = tileN + waveN + j * 16 + rowSel;
            float badd = bias ? bias[ncol] : 0.0f;
            #pragma unroll
            for (int r = 0; r < 8; ++r) {
                int mrow = tileM + waveM + i * 16 + mOff + r;
                C[(size_t)mrow * N + ncol] = acc[i][j][r] + badd;
            }
        }
    }
}

// y[i] = sum_k W[i][k] * x[k] + b[i]
__global__ void gemv_bias_k(const float* __restrict__ W, const float* __restrict__ x,
                            const float* __restrict__ b, float* __restrict__ y,
                            int N, int K)
{
    int i = blockIdx.x * blockDim.x + threadIdx.x;
    if (i >= N) return;
    float s = b[i];
    for (int k = 0; k < K; ++k) s += W[(size_t)i * K + k] * x[k];
    y[i] = s;
}

// split fp32 -> f16 hi/lo planes, 4 elements per thread (n % 4 == 0)
__global__ void split_f16x2_k(const float* __restrict__ x,
                              _Float16* __restrict__ hi, _Float16* __restrict__ lo,
                              long n)
{
    long i = ((long)blockIdx.x * blockDim.x + threadIdx.x) * 4;
    if (i >= n) return;
    float4 v = *(const float4*)(x + i);
    v8h dummy; (void)dummy;
    _Float16 h0 = (_Float16)v.x, h1 = (_Float16)v.y, h2 = (_Float16)v.z, h3 = (_Float16)v.w;
    typedef __attribute__((ext_vector_type(4))) _Float16 v4h;
    v4h vh = { h0, h1, h2, h3 };
    v4h vl = { (_Float16)(v.x - (float)h0), (_Float16)(v.y - (float)h1),
               (_Float16)(v.z - (float)h2), (_Float16)(v.w - (float)h3) };
    *(v4h*)(hi + i) = vh;
    *(v4h*)(lo + i) = vl;
}

extern "C" void kernel_launch(void* const* d_in, const int* in_sizes, int n_in,
                              void* d_out, int out_size, void* d_ws, size_t ws_size,
                              hipStream_t stream)
{
    // softmax over a size-1 axis == 1.0  =>  out = ((vit@Wv.T+bv)@Wiv.T+biv)@Wo.T+bo
    const float* vit = (const float*)d_in[1];
    const float* Wv  = (const float*)d_in[6];   // [E, VITD]
    const float* bv  = (const float*)d_in[7];
    const float* inW = (const float*)d_in[8];   // [3E, E]
    const float* inB = (const float*)d_in[9];
    const float* Wo  = (const float*)d_in[10];  // [E, E]
    const float* bo  = (const float*)d_in[11];
    const float* Wiv = inW + (size_t)2 * E_DIM * E_DIM;
    const float* biv = inB + (size_t)2 * E_DIM;
    float* out = (float*)d_out;

    // workspace layout
    char* ws = (char*)d_ws;
    float*    w1    = (float*)ws;                 ws += (size_t)E_DIM * VITD * 4;   // [E,VITD]
    float*    w2    = (float*)ws;                 ws += (size_t)E_DIM * VITD * 4;   // [E,VITD]
    float*    b1    = (float*)ws;                 ws += E_DIM * 4;
    float*    b2    = (float*)ws;                 ws += E_DIM * 4;
    _Float16* w2hi  = (_Float16*)ws;              ws += (size_t)E_DIM * VITD * 2;   // [N=E][K=VITD]
    _Float16* w2lo  = (_Float16*)ws;              ws += (size_t)E_DIM * VITD * 2;
    _Float16* vithi = (_Float16*)ws;              ws += (size_t)B_ROWS * VITD * 2;  // [M=B][K=VITD]
    _Float16* vitlo = (_Float16*)ws;

    // vit split is independent of the folds — launch first
    {
        long n = (long)B_ROWS * VITD;
        split_f16x2_k<<<dim3((unsigned)(n / 4 / 256)), dim3(256), 0, stream>>>(vit, vithi, vitlo, n);
    }

    // bias folds: b1 = Wiv@bv + biv ; b2 = Wo@b1 + bo
    gemv_bias_k<<<dim3(E_DIM / 256), dim3(256), 0, stream>>>(Wiv, bv, biv, b1, E_DIM, E_DIM);
    gemv_bias_k<<<dim3(E_DIM / 256), dim3(256), 0, stream>>>(Wo, b1, bo, b2, E_DIM, E_DIM);

    // W1 = Wiv @ Wv ; W2 = Wo @ W1
    wmma_gemm_f32x3<<<dim3(E_DIM / 64, VITD / 64), dim3(128), 0, stream>>>(
        Wiv, Wv, nullptr, w1, E_DIM, VITD, E_DIM, VITD, 1);
    wmma_gemm_f32x3<<<dim3(E_DIM / 64, VITD / 64), dim3(128), 0, stream>>>(
        Wo, w1, nullptr, w2, E_DIM, VITD, E_DIM, VITD, 1);

    // split W2 -> f16 hi/lo planes ([N][K] with K contiguous — no transpose needed)
    {
        long n = (long)E_DIM * VITD;
        split_f16x2_k<<<dim3((unsigned)(n / 4 / 256)), dim3(256), 0, stream>>>(w2, w2hi, w2lo, n);
    }

    // out = vit @ W2^T + b2
    wmma_gemm_split<<<dim3(B_ROWS / 64, E_DIM / 64), dim3(128), 0, stream>>>(
        vithi, vitlo, w2hi, w2lo, b2, out, B_ROWS, E_DIM, VITD);
}
